// Model_78426102825621
// MI455X (gfx1250) — matrix operand order repaired and checked by reading, not compiled, not run on hardware
//
#include <hip/hip_runtime.h>

// ---------------------------------------------------------------------------
// Skeleton-GCN forward for MI455X (gfx1250, wave32, WMMA 16x16x32 bf16).
// All heavy channel GEMMs / attention / graph aggregation run on
// v_wmma_f32_16x16x32_bf16 with f32 accumulate; activations stored
// channels-last bf16 so B-fragments & D-stores are contiguous per lane.
// ---------------------------------------------------------------------------

typedef __attribute__((ext_vector_type(16))) __bf16 v16bf;
typedef __attribute__((ext_vector_type(8)))  __bf16 v8bf;
typedef __attribute__((ext_vector_type(8)))  float  v8f;

#define DEVI __device__ __forceinline__

#define NMB 128          // N*M persons-sequences
#define VJ  25           // joints
#define TSEG 64          // frames
#define SP  (NMB*VJ*TSEG)   // 204800 spatial rows
#define NT  (NMB*TSEG)      // 8192 (nm,t) frames
#define S2  (NMB*20)        // 2560 pooled rows

static DEVI v16bf combine16(v8bf lo, v8bf hi) {
  v16bf a;
#pragma unroll
  for (int i = 0; i < 8; i++) { a[i] = lo[i]; a[i + 8] = hi[i]; }
  return a;
}

static DEVI v8f wmma_bf16(v16bf a, v16bf b, v8f c) {
  return __builtin_amdgcn_wmma_f32_16x16x32_bf16(false, a, false, b, (short)0, c,
                                                 false, false);
}

// ---------------------------------------------------------------------------
// BN folding: scale = gamma/sqrt(var+eps), off = beta - mean*scale
// ---------------------------------------------------------------------------
__global__ void k_bn_prep(const float* __restrict__ sc, const float* __restrict__ bi,
                          const float* __restrict__ me, const float* __restrict__ va,
                          float* __restrict__ os, float* __restrict__ oo, int n) {
  int i = blockIdx.x * blockDim.x + threadIdx.x;
  if (i >= n) return;
  float inv = sc[i] * rsqrtf(va[i] + 1e-5f);
  os[i] = inv;
  oo[i] = bi[i] - me[i] * inv;
}

// ---------------------------------------------------------------------------
// Repack f32 row-major weight W[Cout][Cin] into WMMA A-fragment lane order:
// lanes 0-15: m=mt*16+l, K chunks {0..7, 16..23}; lanes 16-31: m=l-16,
// chunks {8..15, 24..31} (16-bit A layout, CDNA5 ISA 7.12.2).
// ---------------------------------------------------------------------------
__global__ void k_pack_w(const float* __restrict__ W, __bf16* __restrict__ out,
                         int Cout, int Cin) {
  int tid = blockIdx.x * blockDim.x + threadIdx.x;
  int KT = Cin >> 5;
  int total = (Cout >> 4) * KT * 32;
  if (tid >= total) return;
  int lane = tid & 31, tile = tid >> 5;
  int kt = tile % KT, mt = tile / KT;
  int m = mt * 16 + (lane & 15);
  int kb = kt * 32 + 8 * (lane >> 4);
  __bf16* dst = out + (size_t)tid * 16;
#pragma unroll
  for (int i = 0; i < 16; i++) {
    int k = kb + (i < 8 ? i : i + 8);
    dst[i] = (__bf16)W[(size_t)m * Cin + k];
  }
}

// ---------------------------------------------------------------------------
// Input permute + motion diff + BN(C*V) + first 1x1 conv (3->64) + ReLU.
// x: (N=64,C=3,T=64,V=25,M=2). Outputs hJ,hD channels-last bf16 [SP x 64].
// ---------------------------------------------------------------------------
__global__ void k_prep(const float* __restrict__ x,
                       const float* __restrict__ invJ, const float* __restrict__ offJ,
                       const float* __restrict__ invD, const float* __restrict__ offD,
                       const float* __restrict__ jw1, const float* __restrict__ jb1,
                       const float* __restrict__ dw1, const float* __restrict__ db1,
                       __bf16* __restrict__ hJ, __bf16* __restrict__ hD) {
  int g = blockIdx.x * blockDim.x + threadIdx.x;
  if (g >= SP * 64) return;
  int o = g & 63, s = g >> 6;
  int t = s % TSEG, v = (s / TSEG) % VJ, nm = s / (TSEG * VJ);
  int n = nm >> 1, m = nm & 1;
  float xs[3], ds[3];
#pragma unroll
  for (int c = 0; c < 3; c++) {
    size_t base = ((((size_t)n * 3 + c) * TSEG + t) * VJ + v) * 2 + m;
    float cur = x[base];
    float d = (t > 0) ? (cur - x[base - VJ * 2]) : 0.f;  // t-stride = V*M
    int ch = c * VJ + v;
    xs[c] = cur * invJ[ch] + offJ[ch];
    ds[c] = d * invD[ch] + offD[ch];
  }
  float aJ = jb1[o] + jw1[o * 3 + 0] * xs[0] + jw1[o * 3 + 1] * xs[1] + jw1[o * 3 + 2] * xs[2];
  float aD = db1[o] + dw1[o * 3 + 0] * ds[0] + dw1[o * 3 + 1] * ds[1] + dw1[o * 3 + 2] * ds[2];
  hJ[g] = (__bf16)fmaxf(aJ, 0.f);
  hD[g] = (__bf16)fmaxf(aD, 0.f);
}

// ---------------------------------------------------------------------------
// spa1 (depends only on (o,v)) and tem1 (only on (o,t)) lookup tables.
// ---------------------------------------------------------------------------
__global__ void k_tables(const float* __restrict__ sw1, const float* __restrict__ sb1,
                         const float* __restrict__ sw2, const float* __restrict__ sb2,
                         const float* __restrict__ tw1, const float* __restrict__ tb1,
                         const float* __restrict__ tw2, const float* __restrict__ tb2,
                         float* __restrict__ spaT /*25x64*/, float* __restrict__ temT /*64x256*/) {
  int g = blockIdx.x * blockDim.x + threadIdx.x;
  if (g < VJ * 64) {
    int v = g / 64, o2 = g % 64;
    float acc = sb2[o2];
    for (int o = 0; o < 64; o++) {
      float h = fmaxf(sw1[o * VJ + v] + sb1[o], 0.f);
      acc += sw2[o2 * 64 + o] * h;
    }
    spaT[v * 64 + o2] = fmaxf(acc, 0.f);
  } else if (g < VJ * 64 + TSEG * 256) {
    int r = g - VJ * 64;
    int t = r / 256, o2 = r % 256;
    float acc = tb2[o2];
    for (int o = 0; o < 64; o++) {
      float h = fmaxf(tw1[o * 64 + t] + tb1[o], 0.f);
      acc += tw2[o2 * 64 + o] * h;
    }
    temT[t * 256 + o2] = fmaxf(acc, 0.f);
  }
}

// Broadcast spa1 table into channels 64..127 of x128.
__global__ void k_fill_spa(const float* __restrict__ spaT, __bf16* __restrict__ x128) {
  int g = blockIdx.x * blockDim.x + threadIdx.x;
  if (g >= SP * 64) return;
  int o = g & 63, s = g >> 6;
  int v = (s / TSEG) % VJ;
  x128[(size_t)s * 128 + 64 + o] = (__bf16)spaT[v * 64 + o];
}

// ---------------------------------------------------------------------------
// Generic WMMA GEMM: out[s][m] = epilogue( sum_c Wp[m][c] * act[s][c] ).
// One wave = one 16-row s-tile, full Cout sweep, B-fragments register-cached.
// Epilogue: +bias, +res_pre (f32), *bnscale+bnoff, ReLU, +res_post (f32);
// writes bf16 (with stride/col offset) and/or f32.
// ---------------------------------------------------------------------------
template <int CIN>
__global__ void __launch_bounds__(256) k_gemm(
    const __bf16* __restrict__ act, const __bf16* __restrict__ wp, int Cout,
    const float* __restrict__ bias, const float* __restrict__ bnsc,
    const float* __restrict__ bnof, const float* __restrict__ respre,
    const float* __restrict__ respost, int relu, __bf16* __restrict__ obf,
    int ostride, int ocol, float* __restrict__ of32, int S) {
  constexpr int KT = CIN / 32;
  int lane = threadIdx.x & 31, wave = threadIdx.x >> 5;
  int s0 = (blockIdx.x * 8 + wave) * 16;
  if (s0 >= S) return;
  int sl = lane & 15, hi = lane >> 4;
  int srow = s0 + sl;
  v16bf bfr[KT];
  {
    const __bf16* bp = act + (size_t)srow * CIN + 8 * hi;
#pragma unroll
    for (int kt = 0; kt < KT; kt++) bfr[kt] = combine16(*(const v8bf*)(bp + kt * 32), *(const v8bf*)(bp + kt * 32 + 16));
  }
  int MT = Cout >> 4;
  int mrow = hi * 8;
  for (int mt = 0; mt < MT; mt++) {
    v8f acc = {};
    const __bf16* ap = wp + ((size_t)mt * KT * 32 + lane) * 16;
#pragma unroll
    for (int kt = 0; kt < KT; kt++) {
      v16bf a = *(const v16bf*)(ap + (size_t)kt * 512);
      acc = wmma_bf16(a, bfr[kt], acc);
    }
    int mb = mt * 16 + mrow;
    float vals[8];
#pragma unroll
    for (int r = 0; r < 8; r++) vals[r] = acc[r];
    if (bias) {
#pragma unroll
      for (int r = 0; r < 8; r++) vals[r] += bias[mb + r];
    }
    if (respre) {
      const float* rp = respre + (size_t)srow * Cout + mb;
#pragma unroll
      for (int r = 0; r < 8; r++) vals[r] += rp[r];
    }
    if (bnsc) {
#pragma unroll
      for (int r = 0; r < 8; r++) vals[r] = vals[r] * bnsc[mb + r] + bnof[mb + r];
    }
    if (relu) {
#pragma unroll
      for (int r = 0; r < 8; r++) vals[r] = fmaxf(vals[r], 0.f);
    }
    if (respost) {
      const float* rp = respost + (size_t)srow * Cout + mb;
#pragma unroll
      for (int r = 0; r < 8; r++) vals[r] += rp[r];
    }
    if (of32) {
      float* op = of32 + (size_t)srow * Cout + mb;
#pragma unroll
      for (int r = 0; r < 8; r++) op[r] = vals[r];
    }
    if (obf) {
      v8bf st;
#pragma unroll
      for (int r = 0; r < 8; r++) st[r] = (__bf16)vals[r];
      *(v8bf*)(obf + (size_t)srow * ostride + ocol + mb) = st;
    }
  }
}

// ---------------------------------------------------------------------------
// Attention: per (nm,t), score[v][u] = sum_c g1[c,v]*g2[c,u]; softmax over u.
// One wave per frame; 2x2 16x16 tiles (V padded to 32), K=256 in 8 WMMA steps.
// Rows/cols >= 25 are clamped on load (garbage discarded) and zeroed in G so
// the GCN stage can do K=32 WMMAs directly. G: [nt][32][32] bf16.
// ---------------------------------------------------------------------------
__global__ void k_attn(const __bf16* __restrict__ g1, const __bf16* __restrict__ g2,
                       __bf16* __restrict__ G) {
  __shared__ float sc[4][32 * 33];  // padded stride 33: conflict-free columns
  int lane = threadIdx.x & 31, wave = threadIdx.x >> 5;
  int nt = blockIdx.x * 4 + wave;
  int nm = nt >> 6, t = nt & 63;
  int vl = lane & 15, hi = lane >> 4;
  int v1 = 16 + vl;
  size_t rowA0 = ((size_t)(nm * VJ + vl) * TSEG + t) * 256;
  size_t rowA1 = ((size_t)(nm * VJ + (v1 < VJ ? v1 : VJ - 1)) * TSEG + t) * 256;
  size_t rowB0 = ((size_t)(nm * VJ + vl) * TSEG + t) * 256;
  size_t rowB1 = rowA1;
  v8f a00 = {}, a01 = {}, a10 = {}, a11 = {};
#pragma unroll
  for (int kt = 0; kt < 8; kt++) {
    int kbA = kt * 32 + 16 * hi;
    int kbB = kt * 32 + 16 * hi;
    v16bf fa0 = combine16(*(const v8bf*)(g1 + rowA0 + kbA),
                          *(const v8bf*)(g1 + rowA0 + kbA + 8));
    v16bf fa1 = combine16(*(const v8bf*)(g1 + rowA1 + kbA),
                          *(const v8bf*)(g1 + rowA1 + kbA + 8));
    v16bf fb0 = *(const v16bf*)(g2 + rowB0 + kbB);
    v16bf fb1 = *(const v16bf*)(g2 + rowB1 + kbB);
    a00 = wmma_bf16(fa0, fb0, a00);
    a01 = wmma_bf16(fa0, fb1, a01);
    a10 = wmma_bf16(fa1, fb0, a10);
    a11 = wmma_bf16(fa1, fb1, a11);
  }
  float* S = sc[wave];
#pragma unroll
  for (int r = 0; r < 8; r++) {
    int row = hi * 8 + r;
    S[row * 33 + vl] = a00[r];
    S[row * 33 + 16 + vl] = a01[r];
    S[(16 + row) * 33 + vl] = a10[r];
    S[(16 + row) * 33 + 16 + vl] = a11[r];
  }
  __syncthreads();
  int v = lane;
  __bf16* gout = G + ((size_t)nt * 32 + v) * 32;
  if (v < VJ) {
    float mx = S[v * 33];
    for (int u = 1; u < VJ; u++) mx = fmaxf(mx, S[v * 33 + u]);
    float sum = 0.f;
    for (int u = 0; u < VJ; u++) {
      float ev = __expf(S[v * 33 + u] - mx);
      S[v * 33 + u] = ev;
      sum += ev;
    }
    float inv = 1.f / sum;
    for (int u = 0; u < 32; u++)
      gout[u] = (__bf16)(u < VJ ? S[v * 33 + u] * inv : 0.f);
  } else {
    for (int u = 0; u < 32; u++) gout[u] = (__bf16)0.f;
  }
}

// ---------------------------------------------------------------------------
// Graph aggregation: Y[v][c] = sum_u G[v][u] * X[u][c] per (nm,t).
// Computed as Y^T = X^T * G^T (M=c, N=v, K=32): X staged u-contiguous in LDS
// (zero-padded u>=25), G rows read contiguous from global; D lands as 8
// consecutive channels -> 16B bf16 stores. One wave per frame.
// ---------------------------------------------------------------------------
template <int C>
__global__ void k_gcn_agg(const __bf16* __restrict__ act, const __bf16* __restrict__ G,
                          __bf16* __restrict__ out) {
  extern __shared__ __bf16 lds[];  // 2 waves * C * 32
  int lane = threadIdx.x & 31, wave = threadIdx.x >> 5;
  int nt = blockIdx.x * 2 + wave;
  int nm = nt >> 6, t = nt & 63;
  __bf16* X = lds + (size_t)wave * C * 32;
  for (int i = lane; i < VJ * C; i += 32) {      // coalesced: c fastest
    int u = i / C, c = i % C;
    X[c * 32 + u] = act[(((size_t)nm * VJ + u) * TSEG + t) * C + c];
  }
  for (int i = lane; i < (32 - VJ) * C; i += 32) {
    int u = VJ + i / C, c = i % C;
    X[c * 32 + u] = (__bf16)0.f;
  }
  __syncthreads();
  int cl = lane & 15, hi = lane >> 4;
  const __bf16* Gb = G + (size_t)nt * 1024;
  v16bf b0 = *(const v16bf*)(Gb + (size_t)cl * 32 + 16 * hi);         // column v=cl
  v16bf b1 = *(const v16bf*)(Gb + (size_t)(16 + cl) * 32 + 16 * hi);  // column v=16+cl
  int v1 = 16 + cl;
#pragma unroll
  for (int ct = 0; ct < (C >> 4); ct++) {
    const __bf16* ap = X + (size_t)(ct * 16 + cl) * 32 + 16 * hi;
    v16bf a = combine16(*(const v8bf*)ap, *(const v8bf*)(ap + 8));
    v8f acc0 = {}, acc1 = {};
    acc0 = wmma_bf16(a, b0, acc0);
    acc1 = wmma_bf16(a, b1, acc1);
    int cb = ct * 16 + hi * 8;
    {
      v8bf st;
#pragma unroll
      for (int r = 0; r < 8; r++) st[r] = (__bf16)acc0[r];
      *(v8bf*)(out + (((size_t)nm * VJ + cl) * TSEG + t) * C + cb) = st;
    }
    if (v1 < VJ) {
      v8bf st;
#pragma unroll
      for (int r = 0; r < 8; r++) st[r] = (__bf16)acc1[r];
      *(v8bf*)(out + (((size_t)nm * VJ + v1) * TSEG + t) * C + cb) = st;
    }
  }
}

// ---------------------------------------------------------------------------
// x3 + tem1 broadcast, max over joints, adaptive max-pool T=64 -> 20.
// P: [NM*20 x 256] channels-last bf16.
// ---------------------------------------------------------------------------
__global__ void k_pool(const __bf16* __restrict__ A3, const float* __restrict__ temT,
                       __bf16* __restrict__ P) {
  int g = blockIdx.x * blockDim.x + threadIdx.x;
  if (g >= NMB * 20 * 256) return;
  int c = g & 255, j = (g >> 8) % 20, nm = g / (20 * 256);
  int ts = (j * TSEG) / 20, te = ((j + 1) * TSEG + 19) / 20;
  float mx = -3.402823466e38f;
  for (int v = 0; v < VJ; v++) {
    const __bf16* base = A3 + (((size_t)nm * VJ + v) * TSEG) * 256 + c;
    for (int t = ts; t < te; t++)
      mx = fmaxf(mx, (float)base[t * 256] + temT[t * 256 + c]);
  }
  P[g] = (__bf16)mx;
}

// conv (1,3) pad 1 over the 20 axis + bias + BN + ReLU
__global__ void k_conv3(const __bf16* __restrict__ P, const float* __restrict__ wc1,
                        const float* __restrict__ bc1, const float* __restrict__ bnsc,
                        const float* __restrict__ bnof, __bf16* __restrict__ O1) {
  int g = blockIdx.x * blockDim.x + threadIdx.x;
  if (g >= NMB * 20 * 256) return;
  int co = g & 255, j = (g >> 8) % 20, nm = g / (20 * 256);
  float acc = bc1[co];
  for (int dj = 0; dj < 3; dj++) {
    int jj = j + dj - 1;
    if (jj < 0 || jj >= 20) continue;
    const __bf16* p = P + ((size_t)nm * 20 + jj) * 256;
    const float* w = wc1 + (size_t)co * 768 + dj;  // wc1[co][ci][dj]
    for (int ci = 0; ci < 256; ci++) acc += w[ci * 3] * (float)p[ci];
  }
  acc = fmaxf(acc * bnsc[co] + bnof[co], 0.f);
  O1[g] = (__bf16)acc;
}

// global max over the 20 pooled frames -> M[nm][512] f32
__global__ void k_maxj(const __bf16* __restrict__ O2, float* __restrict__ M) {
  int g = blockIdx.x * blockDim.x + threadIdx.x;
  if (g >= NMB * 512) return;
  int c = g & 511, nm = g >> 9;
  float mx = -3.402823466e38f;
  for (int j = 0; j < 20; j++)
    mx = fmaxf(mx, (float)O2[((size_t)nm * 20 + j) * 512 + c]);
  M[g] = mx;
}

// mean over the 2 persons + FC 512 -> 60
__global__ void k_fc(const float* __restrict__ M, const float* __restrict__ fw,
                     const float* __restrict__ fb, float* __restrict__ out) {
  int g = blockIdx.x * blockDim.x + threadIdx.x;
  if (g >= 64 * 60) return;
  int k = g % 60, n = g / 60;
  float acc = fb[k];
  const float* m0 = M + (size_t)(2 * n) * 512;
  const float* m1 = M + (size_t)(2 * n + 1) * 512;
  const float* w = fw + (size_t)k * 512;
  for (int c = 0; c < 512; c++) acc += 0.5f * (m0[c] + m1[c]) * w[c];
  out[(size_t)n * 60 + k] = acc;
}

// ---------------------------------------------------------------------------
// Input order: recursive insertion-order flatten of setup_inputs() dict:
// x, then params{joint,dif,spa,tem,g,gcn1,gcn2,gcn3,local,fc}, with
// bnp() = {scale,bias,mean,var}, joint/dif = {bn,w1,b1,w2,b2},
// spa/tem/g = {w1,b1,w2,b2}, gcn = {w,w1,b1,bn}, local = {wc1,bc1,bn1,wc2,bc2,bn2},
// fc = {w,b}.
// ---------------------------------------------------------------------------
enum {
  IN_X = 0,
  J_BN_S, J_BN_B, J_BN_M, J_BN_V, J_W1, J_B1, J_W2, J_B2,
  D_BN_S, D_BN_B, D_BN_M, D_BN_V, D_W1, D_B1, D_W2, D_B2,
  SP_W1, SP_B1, SP_W2, SP_B2,
  TM_W1, TM_B1, TM_W2, TM_B2,
  G_W1, G_B1, G_W2, G_B2,
  G1_W, G1_W1, G1_B1, G1_BN_S, G1_BN_B, G1_BN_M, G1_BN_V,
  G2_W, G2_W1, G2_B1, G2_BN_S, G2_BN_B, G2_BN_M, G2_BN_V,
  G3_W, G3_W1, G3_B1, G3_BN_S, G3_BN_B, G3_BN_M, G3_BN_V,
  L_WC1, L_BC1, L_BN1_S, L_BN1_B, L_BN1_M, L_BN1_V,
  L_WC2, L_BC2, L_BN2_S, L_BN2_B, L_BN2_M, L_BN2_V,
  FC_W, FC_B, IN_COUNT
};

extern "C" void kernel_launch(void* const* d_in, const int* in_sizes, int n_in,
                              void* d_out, int out_size, void* d_ws, size_t ws_size,
                              hipStream_t stream) {
  (void)in_sizes; (void)n_in; (void)out_size; (void)ws_size;
  auto F = [&](int i) { return (const float*)d_in[i]; };

  // ---- workspace carve (deterministic) ----
  char* ws = (char*)d_ws;
  size_t off = 0;
  auto take = [&](size_t bytes) -> char* {
    char* p = ws + off;
    off += (bytes + 255) & ~(size_t)255;
    return p;
  };
  __bf16* Yb   = (__bf16*)take((size_t)SP * 256 * 2);  // agg output / reused for hJ,hD
  __bf16* x128 = (__bf16*)take((size_t)SP * 128 * 2);
  __bf16* g1b  = (__bf16*)take((size_t)SP * 256 * 2);  // reused as A2 after attention
  __bf16* g2b  = (__bf16*)take((size_t)SP * 256 * 2);  // reused as A3 after attention
  __bf16* Gm   = (__bf16*)take((size_t)NT * 1024 * 2);
  float*  Zf   = (float*)take((size_t)SP * 256 * 4);   // f32 partial-sum buffer
  __bf16* A1   = (__bf16*)take((size_t)SP * 128 * 2);
  float*  spaT = (float*)take((size_t)VJ * 64 * 4);
  float*  temT = (float*)take((size_t)TSEG * 256 * 4);
  __bf16* Pb   = (__bf16*)take((size_t)S2 * 256 * 2);
  __bf16* O1   = (__bf16*)take((size_t)S2 * 256 * 2);
  __bf16* O2   = (__bf16*)take((size_t)S2 * 512 * 2);
  float*  Mb   = (float*)take((size_t)NMB * 512 * 4);
  // packed weights
  __bf16* wp_jw2 = (__bf16*)take(64 * 64 * 2);
  __bf16* wp_dw2 = (__bf16*)take(64 * 64 * 2);
  __bf16* wp_gw1 = (__bf16*)take(256 * 128 * 2);
  __bf16* wp_gw2 = (__bf16*)take(256 * 128 * 2);
  __bf16* wp_g1w  = (__bf16*)take(128 * 128 * 2);
  __bf16* wp_g1w1 = (__bf16*)take(128 * 128 * 2);
  __bf16* wp_g2w  = (__bf16*)take(256 * 128 * 2);
  __bf16* wp_g2w1 = (__bf16*)take(256 * 128 * 2);
  __bf16* wp_g3w  = (__bf16*)take(256 * 256 * 2);
  __bf16* wp_g3w1 = (__bf16*)take(256 * 256 * 2);
  __bf16* wp_wc2  = (__bf16*)take(512 * 256 * 2);
  // folded BN params
  float* invJ = (float*)take(75 * 4);  float* offJ = (float*)take(75 * 4);
  float* invD = (float*)take(75 * 4);  float* offD = (float*)take(75 * 4);
  float* b1s = (float*)take(128 * 4);  float* b1o = (float*)take(128 * 4);
  float* b2s = (float*)take(256 * 4);  float* b2o = (float*)take(256 * 4);
  float* b3s = (float*)take(256 * 4);  float* b3o = (float*)take(256 * 4);
  float* l1s = (float*)take(256 * 4);  float* l1o = (float*)take(256 * 4);
  float* l2s = (float*)take(512 * 4);  float* l2o = (float*)take(512 * 4);

  __bf16* hJ = Yb;                       // reuse Y region pre-GCN
  __bf16* hD = Yb + (size_t)SP * 64;
  __bf16* A2 = g1b;
  __bf16* A3 = g2b;

  // ---- parameter prep ----
  auto bnp = [&](int s, int b, int m, int v, float* os, float* oo, int n) {
    k_bn_prep<<<(n + 255) / 256, 256, 0, stream>>>(F(s), F(b), F(m), F(v), os, oo, n);
  };
  bnp(J_BN_S, J_BN_B, J_BN_M, J_BN_V, invJ, offJ, 75);
  bnp(D_BN_S, D_BN_B, D_BN_M, D_BN_V, invD, offD, 75);
  bnp(G1_BN_S, G1_BN_B, G1_BN_M, G1_BN_V, b1s, b1o, 128);
  bnp(G2_BN_S, G2_BN_B, G2_BN_M, G2_BN_V, b2s, b2o, 256);
  bnp(G3_BN_S, G3_BN_B, G3_BN_M, G3_BN_V, b3s, b3o, 256);
  bnp(L_BN1_S, L_BN1_B, L_BN1_M, L_BN1_V, l1s, l1o, 256);
  bnp(L_BN2_S, L_BN2_B, L_BN2_M, L_BN2_V, l2s, l2o, 512);

  auto packw = [&](int wi, __bf16* dst, int Cout, int Cin) {
    int tot = (Cout >> 4) * (Cin >> 5) * 32;
    k_pack_w<<<(tot + 255) / 256, 256, 0, stream>>>(F(wi), dst, Cout, Cin);
  };
  packw(J_W2, wp_jw2, 64, 64);
  packw(D_W2, wp_dw2, 64, 64);
  packw(G_W1, wp_gw1, 256, 128);
  packw(G_W2, wp_gw2, 256, 128);
  packw(G1_W, wp_g1w, 128, 128);
  packw(G1_W1, wp_g1w1, 128, 128);
  packw(G2_W, wp_g2w, 256, 128);
  packw(G2_W1, wp_g2w1, 256, 128);
  packw(G3_W, wp_g3w, 256, 256);
  packw(G3_W1, wp_g3w1, 256, 256);
  packw(L_WC2, wp_wc2, 512, 256);

  k_tables<<<((VJ * 64 + TSEG * 256) + 255) / 256, 256, 0, stream>>>(
      F(SP_W1), F(SP_B1), F(SP_W2), F(SP_B2),
      F(TM_W1), F(TM_B1), F(TM_W2), F(TM_B2), spaT, temT);

  // ---- stage 1: input prep + first conv ----
  k_prep<<<(SP * 64) / 256, 256, 0, stream>>>(F(IN_X), invJ, offJ, invD, offD,
                                              F(J_W1), F(J_B1), F(D_W1), F(D_B1), hJ, hD);

  const int GB = SP / 128;  // 1600 blocks, 8 waves x 16 rows each
  // pos = relu(Wj2*hJ + bj2) -> f32 in Zf
  k_gemm<64><<<GB, 256, 0, stream>>>(hJ, wp_jw2, 64, F(J_B2), nullptr, nullptr,
                                     nullptr, nullptr, 1, nullptr, 0, 0, Zf, SP);
  // dif branch: relu(...) + pos  -> x128 channels 0..63
  k_gemm<64><<<GB, 256, 0, stream>>>(hD, wp_dw2, 64, F(D_B2), nullptr, nullptr,
                                     nullptr, Zf, 1, x128, 128, 0, nullptr, SP);
  // spa1 -> x128 channels 64..127
  k_fill_spa<<<(SP * 64) / 256, 256, 0, stream>>>(spaT, x128);

  // ---- stage 2: attention adjacency ----
  k_gemm<128><<<GB, 256, 0, stream>>>(x128, wp_gw1, 256, F(G_B1), nullptr, nullptr,
                                      nullptr, nullptr, 0, g1b, 256, 0, nullptr, SP);
  k_gemm<128><<<GB, 256, 0, stream>>>(x128, wp_gw2, 256, F(G_B2), nullptr, nullptr,
                                      nullptr, nullptr, 0, g2b, 256, 0, nullptr, SP);
  k_attn<<<NT / 4, 128, 0, stream>>>(g1b, g2b, Gm);

  // ---- stage 3: three GCN layers ----
  // gcn1: 128 -> 128
  k_gcn_agg<128><<<NT / 2, 64, 2 * 128 * 32 * 2, stream>>>(x128, Gm, Yb);
  k_gemm<128><<<GB, 256, 0, stream>>>(Yb, wp_g1w, 128, nullptr, nullptr, nullptr,
                                      nullptr, nullptr, 0, nullptr, 0, 0, Zf, SP);
  k_gemm<128><<<GB, 256, 0, stream>>>(x128, wp_g1w1, 128, F(G1_B1), b1s, b1o,
                                      Zf, nullptr, 1, A1, 128, 0, nullptr, SP);
  // gcn2: 128 -> 256
  k_gcn_agg<128><<<NT / 2, 64, 2 * 128 * 32 * 2, stream>>>(A1, Gm, Yb);
  k_gemm<128><<<GB, 256, 0, stream>>>(Yb, wp_g2w, 256, nullptr, nullptr, nullptr,
                                      nullptr, nullptr, 0, nullptr, 0, 0, Zf, SP);
  k_gemm<128><<<GB, 256, 0, stream>>>(A1, wp_g2w1, 256, F(G2_B1), b2s, b2o,
                                      Zf, nullptr, 1, A2, 256, 0, nullptr, SP);
  // gcn3: 256 -> 256
  k_gcn_agg<256><<<NT / 2, 64, 2 * 256 * 32 * 2, stream>>>(A2, Gm, Yb);
  k_gemm<256><<<GB, 256, 0, stream>>>(Yb, wp_g3w, 256, nullptr, nullptr, nullptr,
                                      nullptr, nullptr, 0, nullptr, 0, 0, Zf, SP);
  k_gemm<256><<<GB, 256, 0, stream>>>(A2, wp_g3w1, 256, F(G3_B1), b3s, b3o,
                                      Zf, nullptr, 1, A3, 256, 0, nullptr, SP);

  // ---- stage 4: +tem1, pooling, local block, head ----
  k_pool<<<(NMB * 20 * 256) / 256, 256, 0, stream>>>(A3, temT, Pb);
  k_conv3<<<(NMB * 20 * 256) / 256, 256, 0, stream>>>(Pb, F(L_WC1), F(L_BC1), l1s, l1o, O1);
  k_gemm<256><<<S2 / 128, 256, 0, stream>>>(O1, wp_wc2, 512, F(L_BC2), l2s, l2o,
                                            nullptr, nullptr, 1, O2, 512, 0, nullptr, S2);
  k_maxj<<<(NMB * 512) / 256, 256, 0, stream>>>(O2, Mb);
  k_fc<<<(64 * 60 + 255) / 256, 256, 0, stream>>>(Mb, F(FC_W), F(FC_B), (float*)d_out);
}
